// TGCNCell_67989332295852
// MI455X (gfx1250) — matrix operand order repaired and checked by reading, not compiled, not run on hardware
//
#include <hip/hip_runtime.h>
#include <hip/hip_bf16.h>

typedef __attribute__((ext_vector_type(16))) _Float16 v16h;
typedef __attribute__((ext_vector_type(8)))  _Float16 v8h;
typedef __attribute__((ext_vector_type(8)))  float    v8f;

#define N_NODES 307
#define U_DIM   100
#define D_INF   3
#define F_INF   103      // U_DIM + D_INF
#define KPAD    128      // K padded to multiple of 32 for wmma f16
#define NEG_SLOPE 0.2f
#define WAVES   8        // waves per gemm block (256 threads, wave32)
#define MAX_NT  13       // max N tiles (208 cols)
#define TPB_G   128      // gather block size

// ---------------- packing kernels (f32 -> padded f16 activations) ----------

// _gc3 input: cat(state[m,0:100], inputs[m,0:3]) -> X[m, 0:128] (pad zeros)
__global__ void pack_gc(const float* __restrict__ state,
                        const float* __restrict__ inputs,
                        _Float16* __restrict__ X, int M) {
  int id = blockIdx.x * blockDim.x + threadIdx.x;
  if (id >= M * KPAD) return;
  int m = id / KPAD, c = id % KPAD;
  float v = 0.f;
  if (c < U_DIM)      v = state[m * U_DIM + c];
  else if (c < F_INF) v = inputs[m * D_INF + (c - U_DIM)];
  X[id] = (_Float16)v;
}

// gru input: cat(inputs[m,0:3], hid[m,0:100]) -> X[m,0:128]
__global__ void pack_gru(const float* __restrict__ inputs,
                         const float* __restrict__ hid,
                         _Float16* __restrict__ X, int M) {
  int id = blockIdx.x * blockDim.x + threadIdx.x;
  if (id >= M * KPAD) return;
  int m = id / KPAD, c = id % KPAD;
  float v = 0.f;
  if (c < D_INF)      v = inputs[m * D_INF + c];
  else if (c < F_INF) v = hid[m * U_DIM + (c - D_INF)];
  X[id] = (_Float16)v;
}

// after gru1: r = sigmoid(g1[:,0:100]+b), u = sigmoid(g1[:,100:200]+b)
// pack X3 = cat(inputs, r*st) and save u
__global__ void pack_x3_u(const float* __restrict__ inputs,
                          const float* __restrict__ g1,      // (M,208)
                          const float* __restrict__ st,
                          const float* __restrict__ b_gru1,  // (200,)
                          float* __restrict__ ubuf,
                          _Float16* __restrict__ X, int M) {
  int id = blockIdx.x * blockDim.x + threadIdx.x;
  if (id >= M * KPAD) return;
  int m = id / KPAD, c = id % KPAD;
  float v = 0.f;
  if (c < D_INF) {
    v = inputs[m * D_INF + c];
  } else if (c < F_INF) {
    int j = c - D_INF;
    float r = 1.f / (1.f + __expf(-(g1[(size_t)m * 208 + j] + b_gru1[j])));
    v = r * st[(size_t)m * U_DIM + j];
    float uu = 1.f / (1.f + __expf(-(g1[(size_t)m * 208 + 100 + j] + b_gru1[100 + j])));
    ubuf[(size_t)m * U_DIM + j] = uu;
  }
  X[id] = (_Float16)v;
}

// weights (F_IN, C) row-major f32 -> Wt[Npad][128] f16, K contiguous per row.
__global__ void conv_w(const float* __restrict__ W0, int C0,
                       const float* __restrict__ W1, int C1,
                       _Float16* __restrict__ Wt, int Npad) {
  int id = blockIdx.x * blockDim.x + threadIdx.x;
  if (id >= Npad * KPAD) return;
  int n = id / KPAD, k = id % KPAD;
  float v = 0.f;
  if (k < F_INF) {
    if (n < C0)           v = W0[k * C0 + n];
    else if (n < C0 + C1) v = W1[k * C1 + (n - C0)];
  }
  Wt[id] = (_Float16)v;
}

// ---------------- WMMA GEMM: Out(M x ldOut) = X(M x 128) @ Wt^T -------------
// All Ntiles (<=13) W tiles staged in LDS once per block; each wave loads its
// A fragments (4 k-steps) once into VGPRs and loops over the N tiles, so the
// 20 MB activation buffer is read exactly once per GEMM.

__global__ void __launch_bounds__(256)
wmma_gemm(const _Float16* __restrict__ X, const _Float16* __restrict__ Wt,
          float* __restrict__ Out, int M, int Ntiles, int ldOut) {
  __shared__ __align__(32) _Float16 sW[MAX_NT * 16 * KPAD];   // up to 53 KB
  const int tid  = threadIdx.x;
  const int lane = tid & 31;
  const int wave = tid >> 5;

  // stage all W tiles (Ntiles * 4KB) through LDS once per block
  {
    const uint32_t* s = (const uint32_t*)Wt;
    uint32_t* d = (uint32_t*)sW;
    const int ndw = Ntiles * 16 * KPAD / 2;
    for (int i = tid; i < ndw; i += 256) d[i] = s[i];
  }
  __syncthreads();

  const int mt = blockIdx.x * WAVES + wave;
  if (mt * 16 >= M) return;                     // wave-uniform, EXEC stays full

  const int g   = lane >> 4;                    // lane half-group
  const int l15 = lane & 15;
  const int m   = mt * 16 + l15;                // A row for this lane

  // A 16x32 f16 fragments for all 4 K-steps (contiguous k order, same as B):
  // lanes 0-15 hold K = k0+[0..15]; lanes 16-31 hold K = k0+16+[0..15].
  v16h a[4];
#pragma unroll
  for (int ks = 0; ks < 4; ++ks) {
    const int k0 = ks * 32;
    v8h lo = *(const v8h*)(X + (size_t)m * KPAD + k0 + g * 16);
    v8h hi = *(const v8h*)(X + (size_t)m * KPAD + k0 + g * 16 + 8);
#pragma unroll
    for (int i = 0; i < 8; ++i) { a[ks][i] = lo[i]; a[ks][i + 8] = hi[i]; }
  }

  for (int nt = 0; nt < Ntiles; ++nt) {
    v8f acc = {};
#pragma unroll
    for (int ks = 0; ks < 4; ++ks) {
      // B 32x16 f16: lane n = l15, K = k0 + g*16 + [0..15] contiguous in LDS
      v16h b = *(const v16h*)(sW + (nt * 16 + l15) * KPAD + ks * 32 + g * 16);
      acc = __builtin_amdgcn_wmma_f32_16x16x32_f16(false, a[ks], false, b,
                                                   (short)0, acc, false, false);
    }
    // C/D: VGPR r -> row M = r + 8*g, col N = l15
#pragma unroll
    for (int r = 0; r < 8; ++r)
      Out[(size_t)(mt * 16 + r + 8 * g) * ldOut + nt * 16 + l15] = acc[r];
  }
}

// ---------------- CSR build over destination nodes (graph is batch-invariant)

__global__ void fill_i32(int* __restrict__ p, int v, int n) {
  int id = blockIdx.x * blockDim.x + threadIdx.x;
  if (id < n) p[id] = v;
}

__global__ void csr_count(const int* __restrict__ dst, int* __restrict__ cnt,
                          int E, int E2) {
  int e = blockIdx.x * blockDim.x + threadIdx.x;
  if (e >= E2) return;
  int d = (e < E) ? dst[e] : (e - E);       // self loops appended
  atomicAdd(&cnt[d], 1);
}

__global__ void csr_scan(const int* __restrict__ cnt, int* __restrict__ rowptr,
                         int* __restrict__ ofs) {
  if (blockIdx.x == 0 && threadIdx.x == 0) {
    int s = 0;
    for (int i = 0; i < N_NODES; ++i) { rowptr[i] = s; ofs[i] = s; s += cnt[i]; }
    rowptr[N_NODES] = s;
  }
}

__global__ void csr_fill(const int* __restrict__ src, const int* __restrict__ dst,
                         int* __restrict__ ofs, int* __restrict__ esrc,
                         int E, int E2) {
  int e = blockIdx.x * blockDim.x + threadIdx.x;
  if (e >= E2) return;
  int s = (e < E) ? src[e] : (e - E);
  int d = (e < E) ? dst[e] : (e - E);
  int pos = atomicAdd(&ofs[d], 1);
  esrc[pos] = s;
}

// ---------------- GATv2 attention as atomic-free gather --------------------
// One block per (b, dst-node). Chunked online softmax over incoming edges:
// per-chunk LDS max/sum tree reductions, rescaled running numerator held in
// registers (thread t owns output channel u = t).

__global__ void __launch_bounds__(TPB_G)
gat_gather(const float* __restrict__ xlr,       // (M,208): xl | xr
           const int* __restrict__ rowptr, const int* __restrict__ esrc,
           const float* __restrict__ att,
           const float* __restrict__ gat_bias, const float* __restrict__ bias_gc,
           float* __restrict__ st, int B) {
  const int bn = blockIdx.x;                    // b*N_NODES + n
  const int n  = bn % N_NODES;
  const int b  = bn / N_NODES;
  const int t  = threadIdx.x;

  __shared__ float att_s[U_DIM];
  __shared__ float xr_s[U_DIM];
  __shared__ float ex_s[TPB_G];
  __shared__ int   src_s[TPB_G];
  __shared__ float red[TPB_G];

  if (t < U_DIM) {
    att_s[t] = att[t];
    xr_s[t]  = xlr[(size_t)bn * 208 + 100 + t];
  }
  __syncthreads();

  const int e0 = rowptr[n], e1 = rowptr[n + 1];
  float runmax = -__builtin_inff();
  float denom  = 0.f;
  float num    = 0.f;                           // valid for t < U_DIM

  for (int base = e0; base < e1; base += TPB_G) {
    const int cnt = min(TPB_G, e1 - base);

    // e_t = att . leaky_relu(xl[src] + xr[n])
    float ev = -__builtin_inff();
    if (t < cnt) {
      int s = esrc[base + t];
      src_s[t] = s;
      const float* xl = xlr + (size_t)(b * N_NODES + s) * 208;
      float acc = 0.f;
#pragma unroll 4
      for (int u = 0; u < U_DIM; ++u) {
        float v = xl[u] + xr_s[u];
        v = (v > 0.f) ? v : NEG_SLOPE * v;
        acc += att_s[u] * v;
      }
      ev = acc;
    }

    // chunk max
    red[t] = ev; __syncthreads();
    for (int o = TPB_G / 2; o > 0; o >>= 1) {
      if (t < o) red[t] = fmaxf(red[t], red[t + o]);
      __syncthreads();
    }
    const float newmax = fmaxf(runmax, red[0]);
    __syncthreads();

    // chunk exp + sum
    float ex = (t < cnt) ? __expf(ev - newmax) : 0.f;
    ex_s[t] = ex;
    red[t] = ex; __syncthreads();
    for (int o = TPB_G / 2; o > 0; o >>= 1) {
      if (t < o) red[t] += red[t + o];
      __syncthreads();
    }
    const float scale = __expf(runmax - newmax);   // 0 on first chunk
    denom = denom * scale + red[0];
    num  *= scale;

    // numerator: coalesced across u = t
    if (t < U_DIM) {
      for (int j = 0; j < cnt; ++j)
        num += ex_s[j] * xlr[(size_t)(b * N_NODES + src_s[j]) * 208 + t];
    }
    runmax = newmax;
    __syncthreads();
  }

  if (t < U_DIM)
    st[(size_t)bn * U_DIM + t] = num / denom + gat_bias[t] + bias_gc[t];
}

// out = u*st + (1-u)*tanh(g2 + b_gru2)
__global__ void final_out(const float* __restrict__ ubuf, const float* __restrict__ st,
                          const float* __restrict__ g2,   // (M,112)
                          const float* __restrict__ b_gru2,
                          float* __restrict__ out, int M) {
  int id = blockIdx.x * blockDim.x + threadIdx.x;
  if (id >= M * U_DIM) return;
  int m = id / U_DIM, u = id % U_DIM;
  float c = tanhf(g2[(size_t)m * 112 + u] + b_gru2[u]);
  float uu = ubuf[id];
  out[id] = uu * st[id] + (1.f - uu) * c;
}

// ---------------------------------------------------------------------------

extern "C" void kernel_launch(void* const* d_in, const int* in_sizes, int n_in,
                              void* d_out, int out_size, void* d_ws, size_t ws_size,
                              hipStream_t stream) {
  const float* inputs   = (const float*)d_in[0];
  const float* state    = (const float*)d_in[1];
  const int*   src      = (const int*)d_in[2];
  const int*   dst      = (const int*)d_in[3];
  const float* Wl       = (const float*)d_in[4];
  const float* Wr       = (const float*)d_in[5];
  const float* att      = (const float*)d_in[6];
  const float* gat_bias = (const float*)d_in[7];
  const float* bias_gc  = (const float*)d_in[8];
  const float* W_gru1   = (const float*)d_in[9];
  const float* b_gru1   = (const float*)d_in[10];
  const float* W_gru2   = (const float*)d_in[11];
  const float* b_gru2   = (const float*)d_in[12];

  const int B  = in_sizes[1] / (N_NODES * U_DIM);   // 256
  const int E  = in_sizes[2];                       // 4096
  const int E2 = E + N_NODES;                       // + self loops
  const int M  = B * N_NODES;                       // 78592 (multiple of 16)

  // workspace layout
  auto al = [](size_t x) { return (x + 255) & ~(size_t)255; };
  char* ws = (char*)d_ws;
  size_t off = 0;
  _Float16* X  = (_Float16*)(ws + off); off = al(off + (size_t)M * KPAD * 2);
  _Float16* Wt = (_Float16*)(ws + off); off = al(off + (size_t)208 * KPAD * 2);
  float* G0   = (float*)(ws + off); off = al(off + (size_t)M * 208 * 4);
  float* ST   = (float*)(ws + off); off = al(off + (size_t)M * U_DIM * 4);
  float* UB   = (float*)(ws + off); off = al(off + (size_t)M * U_DIM * 4);
  int* CNT    = (int*)(ws + off);   off = al(off + (size_t)N_NODES * 4);
  int* ROWPTR = (int*)(ws + off);   off = al(off + (size_t)(N_NODES + 1) * 4);
  int* OFS    = (int*)(ws + off);   off = al(off + (size_t)N_NODES * 4);
  int* ESRC   = (int*)(ws + off);   off = al(off + (size_t)E2 * 4);
  (void)ws_size; (void)n_in; (void)out_size;

  const int TPB = 256;
  const int mtBlocks = (M / 16 + WAVES - 1) / WAVES;  // 614
  const int xGrid = (M * KPAD + TPB - 1) / TPB;

  // ---- CSR over dst (graph identical for all batch samples) ----
  fill_i32<<<(N_NODES + TPB - 1) / TPB, TPB, 0, stream>>>(CNT, 0, N_NODES);
  csr_count<<<(E2 + TPB - 1) / TPB, TPB, 0, stream>>>(dst, CNT, E, E2);
  csr_scan<<<1, 32, 0, stream>>>(CNT, ROWPTR, OFS);
  csr_fill<<<(E2 + TPB - 1) / TPB, TPB, 0, stream>>>(src, dst, OFS, ESRC, E, E2);

  // ---- GATv2: xlr = cat(state,inputs) @ [Wl|Wr] ----
  pack_gc<<<xGrid, TPB, 0, stream>>>(state, inputs, X, M);
  conv_w<<<(208 * KPAD + TPB - 1) / TPB, TPB, 0, stream>>>(Wl, U_DIM, Wr, U_DIM, Wt, 208);
  wmma_gemm<<<mtBlocks, TPB, 0, stream>>>(X, Wt, G0, M, 13, 208);

  // ---- edge softmax + aggregate, atomic-free gather ----
  gat_gather<<<M, TPB_G, 0, stream>>>(G0, ROWPTR, ESRC, att, gat_bias, bias_gc, ST, B);

  // ---- GRU gate 1: sigmoid(cat(inputs, st) @ W_gru1 + b) ----
  pack_gru<<<xGrid, TPB, 0, stream>>>(inputs, ST, X, M);
  conv_w<<<(208 * KPAD + TPB - 1) / TPB, TPB, 0, stream>>>(W_gru1, 2 * U_DIM, W_gru1, 0, Wt, 208);
  wmma_gemm<<<mtBlocks, TPB, 0, stream>>>(X, Wt, G0, M, 13, 208);

  // ---- candidate: c = tanh(cat(inputs, r*st) @ W_gru2 + b) ----
  pack_x3_u<<<xGrid, TPB, 0, stream>>>(inputs, G0, ST, b_gru1, UB, X, M);
  conv_w<<<(112 * KPAD + TPB - 1) / TPB, TPB, 0, stream>>>(W_gru2, U_DIM, W_gru2, 0, Wt, 112);
  wmma_gemm<<<mtBlocks, TPB, 0, stream>>>(X, Wt, G0, M, 7, 112);

  // ---- out = u*st + (1-u)*c ----
  final_out<<<(M * U_DIM + TPB - 1) / TPB, TPB, 0, stream>>>(
      UB, ST, G0, b_gru2, (float*)d_out, M);
}
